// GRUDirectModel_29652454212320
// MI455X (gfx1250) — matrix operand order repaired and checked by reading, not compiled, not run on hardware
//
#include <hip/hip_runtime.h>
#include <hip/hip_bf16.h>
#include <stdint.h>

#define B_  512
#define T_  336
#define E_  16
#define H_  128
#define FH_ 24
#define FT_ 4
#define G3_ 384   /* 3*H */

typedef __attribute__((ext_vector_type(16))) __bf16 v16bf;
typedef __attribute__((ext_vector_type(8)))  float  v8f;

/* ---------------- LDS layout (bytes) ----------------
   Three weight matrices pre-swizzled into WMMA B-fragment order:
   frag(nt,kt) = 1024B = 32 lanes x 8 dwords; lane = N col (lane&15), half=lane>>4,
   dword v = packed bf16 (K=8*half+2v+(v>=4?8:0) low, K+1 high).
   h0/h1/xA are ping-pong double-buffered so each GRU sub-step needs
   only ONE barrier (read cur / write nxt are disjoint).              */
#define WHH0_OFF 0
#define WIH1_OFF 98304
#define WHH1_OFF 196608
#define H0_OFF0  294912   /* 16 x 136 bf16 (row stride 272B, bank-skewed) */
#define H0_OFF1  299264
#define H1_OFF0  303616
#define H1_OFF1  307968
#define XA_OFF0  312320   /* 16 x 40 bf16, K cols 16..31 stay zero */
#define XA_OFF1  313600
#define BI0_OFF  314880
#define BH0_OFF  316416
#define BI1_OFF  317952
#define BH1_OFF  319488
#define SMEM_BYTES 321024

#define HRS 272
#define XRS 80

__device__ __forceinline__ uint32_t f2bf(float f){
  union { float f; uint32_t u; } c; c.f = f;
  uint32_t u = c.u;
  u += 0x7FFFu + ((u >> 16) & 1u);      /* round-to-nearest-even */
  return u >> 16;
}
__device__ __forceinline__ float bf2f(uint16_t b){
  union { uint32_t u; float f; } c; c.u = ((uint32_t)b) << 16;
  return c.f;
}

union Frag {
  v16bf    v;
  uint32_t u[8];
  uint64_t q[4];
};

__device__ __forceinline__ v8f vzero(){
  v8f z = {0.f,0.f,0.f,0.f,0.f,0.f,0.f,0.f};
  return z;
}

/* A-fragment (16-bit, 16x32): base already includes m*rowstride + half*16B.
   v0..3 <- K = kt*32 + half*8 + 0..7 ; v4..7 <- +16.                      */
__device__ __forceinline__ void loadA(Frag& a, const char* base, int ktb){
  a.q[0] = *(const uint64_t*)(base + ktb + 0);
  a.q[1] = *(const uint64_t*)(base + ktb + 8);
  a.q[2] = *(const uint64_t*)(base + ktb + 32);
  a.q[3] = *(const uint64_t*)(base + ktb + 40);
}
__device__ __forceinline__ void loadB(Frag& b, const char* base){
  b.q[0] = ((const uint64_t*)base)[0];
  b.q[1] = ((const uint64_t*)base)[1];
  b.q[2] = ((const uint64_t*)base)[2];
  b.q[3] = ((const uint64_t*)base)[3];
}

#define WMMA_BF16(A, Bf, C) \
  __builtin_amdgcn_wmma_f32_16x16x32_bf16(false, (A).v, false, (Bf).v, (short)0, (C), false, false)

/* Swizzle a [384][128] f32 weight into B-fragment bf16 layout in LDS. */
__device__ __forceinline__ void swizzleW(const float* __restrict__ W, char* dst, int tid){
  for (int idx = tid; idx < 24*4*32*8; idx += 256){
    int v    = idx & 7;
    int lane = (idx >> 3) & 31;
    int kt   = (idx >> 8) & 3;
    int nt   = idx >> 10;
    int k = kt*32 + 8*(lane >> 4) + 2*v + ((v >= 4) ? 8 : 0);
    int g = nt*16 + (lane & 15);
    uint32_t lo = f2bf(W[g*H_ + k]);
    uint32_t hi = f2bf(W[g*H_ + k + 1]);
    ((uint32_t*)dst)[idx] = lo | (hi << 16);
  }
}

/* fast activations: single v_rcp_f32 instead of IEEE div chains */
__device__ __forceinline__ float frcp(float x){ return __builtin_amdgcn_rcpf(x); }
__device__ __forceinline__ float sigm(float x){ return frcp(1.f + __expf(-x)); }
__device__ __forceinline__ float tanh_fast(float x){
  /* tanh(x) = 1 - 2/(exp(2x)+1); saturates correctly at +/-inf */
  return 1.f - 2.f * frcp(1.f + __expf(2.f * x));
}

__global__ __launch_bounds__(256) void gru_fused(
    const float* __restrict__ x_enc,
    const float* __restrict__ W_ih0, const float* __restrict__ W_hh0,
    const float* __restrict__ b_ih0, const float* __restrict__ b_hh0,
    const float* __restrict__ W_ih1, const float* __restrict__ W_hh1,
    const float* __restrict__ b_ih1, const float* __restrict__ b_hh1,
    float* __restrict__ h_last)
{
  extern __shared__ char smem[];
  const int tid  = threadIdx.x;
  const int lane = tid & 31;
  const int wv   = tid >> 5;          /* 8 wave32s */
  const int bb   = blockIdx.x << 4;   /* 16 batch rows per WG */

  /* ---- init: zero h0/h1/xA (both buffers), biases, weight swizzle ---- */
  for (int i = tid; i < (BI0_OFF - H0_OFF0)/4; i += 256)
    ((uint32_t*)(smem + H0_OFF0))[i] = 0u;
  {
    float* bi0p = (float*)(smem + BI0_OFF);
    float* bh0p = (float*)(smem + BH0_OFF);
    float* bi1p = (float*)(smem + BI1_OFF);
    float* bh1p = (float*)(smem + BH1_OFF);
    for (int i = tid; i < G3_; i += 256){
      bi0p[i] = b_ih0[i]; bh0p[i] = b_hh0[i];
      bi1p[i] = b_ih1[i]; bh1p[i] = b_hh1[i];
    }
  }
  swizzleW(W_hh0, smem + WHH0_OFF, tid);
  swizzleW(W_ih1, smem + WIH1_OFF, tid);
  swizzleW(W_hh1, smem + WHH1_OFF, tid);

  /* W_ih0 fragments (K=16, zero-padded to 32) in registers per wave */
  const int nt0 = wv, nt1 = wv + 8, nt2 = wv + 16;
  Frag wih0[3];
  {
    const int nts[3] = {nt0, nt1, nt2};
    for (int j = 0; j < 3; ++j){
      #pragma unroll
      for (int v = 0; v < 8; ++v){
        uint32_t p = 0u;
        if (v < 4){
          int g = nts[j]*16 + (lane & 15);
          int k0 = 8*(lane >> 4) + 2*v;
          p = f2bf(W_ih0[g*E_ + k0]) | (f2bf(W_ih0[g*E_ + k0 + 1]) << 16);
        }
        wih0[j].u[v] = p;
      }
    }
  }

  const int xm = tid >> 4, xe = tid & 15;
  const float* xrow = x_enc + (size_t)(bb + xm)*T_*E_ + xe;

  /* integer byte-offset tables (NOT pointer arrays: LDS addrspacecast
     cannot appear in a static initializer) */
  const int m    = lane & 15;
  const int half = lane >> 4;
  const int h0Aoff[2] = { H0_OFF0 + m*HRS + half*16, H0_OFF1 + m*HRS + half*16 };
  const int h1Aoff[2] = { H1_OFF0 + m*HRS + half*16, H1_OFF1 + m*HRS + half*16 };
  const int xaAoff[2] = { XA_OFF0 + m*XRS + half*16, XA_OFF1 + m*XRS + half*16 };
  const int h0Soff[2] = { H0_OFF0, H0_OFF1 };
  const int h1Soff[2] = { H1_OFF0, H1_OFF1 };
  const int xaSoff[2] = { XA_OFF0, XA_OFF1 };

  /* stage x_0 into buffer 0 before the init barrier */
  ((uint16_t*)(smem + XA_OFF0))[xm*40 + xe] = (uint16_t)f2bf(xrow[0]);
  __syncthreads();

  const int   gc = (wv << 4) + (lane & 15);   /* gate column in [0,128) */
  const float* bi0 = (const float*)(smem + BI0_OFF);
  const float* bh0 = (const float*)(smem + BH0_OFF);
  const float* bi1 = (const float*)(smem + BI1_OFF);
  const float* bh1 = (const float*)(smem + BH1_OFF);
  const float bi0r = bi0[gc], bi0z = bi0[128+gc], bi0n = bi0[256+gc];
  const float bh0r = bh0[gc], bh0z = bh0[128+gc], bh0n = bh0[256+gc];
  const float bi1r = bi1[gc], bi1z = bi1[128+gc], bi1n = bi1[256+gc];
  const float bh1r = bh1[gc], bh1z = bh1[128+gc], bh1n = bh1[256+gc];

  for (int t = 0; t < T_; ++t){
    const int cur = t & 1, nxt = cur ^ 1;

    /* stage x_{t+1} into the other buffer; consumed after next barrier */
    if (t + 1 < T_){
      ((uint16_t*)(smem + xaSoff[nxt]))[xm*40 + xe] = (uint16_t)f2bf(xrow[(t+1)*E_]);
      __builtin_prefetch(xrow + (t+2)*E_, 0, 0);
    }

    /* ---------- layer 0: gh = h0 @ Whh0^T ; xp = x_t @ Wih0^T ---------- */
    v8f g0 = vzero(), g1 = vzero(), g2 = vzero();
    {
      const char* aBase = smem + h0Aoff[cur];
      #pragma unroll
      for (int kt = 0; kt < 4; ++kt){
        Frag a; loadA(a, aBase, kt*64);
        Frag bf;
        loadB(bf, smem + WHH0_OFF + ((nt0*4 + kt) << 10) + lane*32);
        g0 = WMMA_BF16(a, bf, g0);
        loadB(bf, smem + WHH0_OFF + ((nt1*4 + kt) << 10) + lane*32);
        g1 = WMMA_BF16(a, bf, g1);
        loadB(bf, smem + WHH0_OFF + ((nt2*4 + kt) << 10) + lane*32);
        g2 = WMMA_BF16(a, bf, g2);
      }
    }
    v8f x0 = vzero(), x1 = vzero(), x2 = vzero();
    {
      Frag a; loadA(a, smem + xaAoff[cur], 0);
      x0 = WMMA_BF16(a, wih0[0], x0);
      x1 = WMMA_BF16(a, wih0[1], x1);
      x2 = WMMA_BF16(a, wih0[2], x2);
    }
    {
      const uint16_t* h0c = (const uint16_t*)(smem + h0Soff[cur]);
      uint16_t*       h0n = (uint16_t*)(smem + h0Soff[nxt]);
      #pragma unroll
      for (int i = 0; i < 8; ++i){
        int mr = (half << 3) + i;
        float r  = sigm(x0[i] + bi0r + g0[i] + bh0r);
        float z  = sigm(x1[i] + bi0z + g1[i] + bh0z);
        float nn = tanh_fast(x2[i] + bi0n + r*(g2[i] + bh0n));
        float hp = bf2f(h0c[mr*136 + gc]);
        h0n[mr*136 + gc] = (uint16_t)f2bf((1.f - z)*nn + z*hp);
      }
    }
    __syncthreads();                       /* h0[nxt] ready */

    /* ---------- layer 1: gh = h1 @ Whh1^T ; xp = h0_new @ Wih1^T ------- */
    g0 = vzero(); g1 = vzero(); g2 = vzero();
    x0 = vzero(); x1 = vzero(); x2 = vzero();
    {
      const char* ahBase = smem + h1Aoff[cur];
      const char* axBase = smem + h0Aoff[nxt];
      #pragma unroll
      for (int kt = 0; kt < 4; ++kt){
        Frag ah; loadA(ah, ahBase, kt*64);
        Frag ax; loadA(ax, axBase, kt*64);
        Frag bf;
        loadB(bf, smem + WHH1_OFF + ((nt0*4 + kt) << 10) + lane*32);
        g0 = WMMA_BF16(ah, bf, g0);
        loadB(bf, smem + WHH1_OFF + ((nt1*4 + kt) << 10) + lane*32);
        g1 = WMMA_BF16(ah, bf, g1);
        loadB(bf, smem + WHH1_OFF + ((nt2*4 + kt) << 10) + lane*32);
        g2 = WMMA_BF16(ah, bf, g2);
        loadB(bf, smem + WIH1_OFF + ((nt0*4 + kt) << 10) + lane*32);
        x0 = WMMA_BF16(ax, bf, x0);
        loadB(bf, smem + WIH1_OFF + ((nt1*4 + kt) << 10) + lane*32);
        x1 = WMMA_BF16(ax, bf, x1);
        loadB(bf, smem + WIH1_OFF + ((nt2*4 + kt) << 10) + lane*32);
        x2 = WMMA_BF16(ax, bf, x2);
      }
    }
    {
      const uint16_t* h1c = (const uint16_t*)(smem + h1Soff[cur]);
      uint16_t*       h1n = (uint16_t*)(smem + h1Soff[nxt]);
      #pragma unroll
      for (int i = 0; i < 8; ++i){
        int mr = (half << 3) + i;
        float r  = sigm(x0[i] + bi1r + g0[i] + bh1r);
        float z  = sigm(x1[i] + bi1z + g1[i] + bh1z);
        float nn = tanh_fast(x2[i] + bi1n + r*(g2[i] + bh1n));
        float hp = bf2f(h1c[mr*136 + gc]);
        h1n[mr*136 + gc] = (uint16_t)f2bf((1.f - z)*nn + z*hp);
      }
    }
    __syncthreads();                       /* h1[nxt] ready */
  }

  /* final h1 is in buffer (T_ & 1) == 0 ; dump f32 to workspace */
  {
    const uint16_t* hf = (const uint16_t*)(smem + h1Soff[T_ & 1]);
    for (int idx = tid; idx < 16*H_; idx += 256){
      int mm = idx >> 7, k = idx & 127;
      h_last[(size_t)(bb + mm)*H_ + k] = bf2f(hf[mm*136 + k]);
    }
  }
}

/* ---- MLP head: relu(fused @ W1^T + b1) @ W2^T + b2, fused=[h|xft] ---- */
__global__ __launch_bounds__(128) void mlp_head(
    const float* __restrict__ hlast, const float* __restrict__ xft,
    const float* __restrict__ W1, const float* __restrict__ b1,
    const float* __restrict__ W2, const float* __restrict__ b2,
    float* __restrict__ out)
{
  const int b = blockIdx.x;
  const int k = threadIdx.x;          /* 128 threads = hidden units */
  __shared__ float sh[128];
  __shared__ float sxf[FH_*FT_];
  __shared__ float red[128];

  sh[k] = hlast[(size_t)b*H_ + k];
  if (k < FH_*FT_) sxf[k] = xft[(size_t)b*FH_*FT_ + k];
  __syncthreads();

  const float* w1r = W1 + (size_t)k*(H_ + FT_);
  float g = b1[k];
  #pragma unroll 8
  for (int d = 0; d < H_; ++d) g += sh[d] * w1r[d];
  const float w2k = W2[k];
  const float b2v = b2[0];

  for (int f = 0; f < FH_; ++f){
    float a = g + w1r[128]*sxf[f*4+0] + w1r[129]*sxf[f*4+1]
                + w1r[130]*sxf[f*4+2] + w1r[131]*sxf[f*4+3];
    a = fmaxf(a, 0.f);
    red[k] = a * w2k;
    __syncthreads();
    for (int s = 64; s > 0; s >>= 1){
      if (k < s) red[k] += red[k + s];
      __syncthreads();
    }
    if (k == 0) out[(size_t)b*FH_ + f] = red[0] + b2v;
    __syncthreads();
  }
}

extern "C" void kernel_launch(void* const* d_in, const int* in_sizes, int n_in,
                              void* d_out, int out_size, void* d_ws, size_t ws_size,
                              hipStream_t stream)
{
  const float* x_enc = (const float*)d_in[0];
  const float* xft   = (const float*)d_in[1];
  const float* W_ih0 = (const float*)d_in[2];
  const float* W_hh0 = (const float*)d_in[3];
  const float* b_ih0 = (const float*)d_in[4];
  const float* b_hh0 = (const float*)d_in[5];
  const float* W_ih1 = (const float*)d_in[6];
  const float* W_hh1 = (const float*)d_in[7];
  const float* b_ih1 = (const float*)d_in[8];
  const float* b_hh1 = (const float*)d_in[9];
  const float* W1    = (const float*)d_in[10];
  const float* b1    = (const float*)d_in[11];
  const float* W2    = (const float*)d_in[12];
  const float* b2    = (const float*)d_in[13];
  float* out   = (float*)d_out;
  float* hlast = (float*)d_ws;        /* 512*128*4 = 256 KB */

  (void)hipFuncSetAttribute(reinterpret_cast<const void*>(gru_fused),
                            hipFuncAttributeMaxDynamicSharedMemorySize, SMEM_BYTES);

  gru_fused<<<B_/16, 256, SMEM_BYTES, stream>>>(
      x_enc, W_ih0, W_hh0, b_ih0, b_hh0,
      W_ih1, W_hh1, b_ih1, b_hh1, hlast);

  mlp_head<<<B_, 128, 0, stream>>>(hlast, xft, W1, b1, W2, b2, out);
}
